// Model_3444563771826
// MI455X (gfx1250) — compile-verified, matrix operand order checked by reading, not run on hardware
//
#include <hip/hip_runtime.h>
#include <hip/hip_bf16.h>

// ---------------------------------------------------------------------------
// GPT forward for gfx1250 (MI455X).
// All GEMMs: bf16 operands, v_wmma_f32_16x16x32_bf16, f32 accumulate.
// Tile staging: global_load_async_to_lds_b128 (ASYNCcnt) double-buffered,
// with a synchronous fallback if the builtin is unavailable.
// Weights are convert+transposed to bf16 [N,K] once per launch (pre-pass).
// ---------------------------------------------------------------------------

typedef __attribute__((ext_vector_type(16))) __bf16 v16bf;
typedef __attribute__((ext_vector_type(8)))  __bf16 v8bf;
typedef __attribute__((ext_vector_type(8)))  float  v8f;
typedef __attribute__((ext_vector_type(4)))  int    v4i;

// pointer-to-v4i in explicit address spaces (for the async-copy builtin)
typedef __attribute__((address_space(1))) v4i* gv4i_p;   // global
typedef __attribute__((address_space(3))) v4i* lv4i_p;   // LDS

#define BM 128
#define BN 64
#define BK 32
#define BKP 40   // padded LDS row stride in bf16 elems (80B: 16B aligned, bank-spread)

#if defined(__has_builtin)
# if __has_builtin(__builtin_amdgcn_global_load_async_to_lds_b128)
#  define ASYNC_COPY 1
# endif
#endif
#ifndef ASYNC_COPY
# define ASYNC_COPY 0
#endif

__device__ inline __bf16 cvt_bf16(float f) {
  union { float f; unsigned u; } v; v.f = f;
  unsigned r = v.u + 0x7FFFu + ((v.u >> 16) & 1u);   // round-to-nearest-even
  unsigned short h = (unsigned short)(r >> 16);
  __bf16 out; __builtin_memcpy(&out, &h, 2);
  return out;
}

// 16-byte global -> LDS copy (per lane)
__device__ inline void async_cp16(const __bf16* g, __bf16* l) {
#if ASYNC_COPY
  __builtin_amdgcn_global_load_async_to_lds_b128((gv4i_p)g, (lv4i_p)l, 0, 0);
#else
  *(v8bf*)l = *(const v8bf*)g;
#endif
}

__device__ inline void wait_async_pending() {   // allow 3 in flight (next tile)
#if ASYNC_COPY
# if __has_builtin(__builtin_amdgcn_s_wait_asynccnt)
  __builtin_amdgcn_s_wait_asynccnt(3);
# else
  asm volatile("s_wait_asynccnt 0x3" ::: "memory");
# endif
#endif
}
__device__ inline void wait_async_all() {
#if ASYNC_COPY
# if __has_builtin(__builtin_amdgcn_s_wait_asynccnt)
  __builtin_amdgcn_s_wait_asynccnt(0);
# else
  asm volatile("s_wait_asynccnt 0x0" ::: "memory");
# endif
#endif
}

// Generic batched GEMM: C[z] = act( scale * A[z] @ B[z]^T (+bias) (+Radd) )
//   A: [M,K] bf16 row-major (lda), batch stride sAz
//   B: [N,K] bf16 row-major (ldb), batch stride sBz
//   C index: base = (z/zDiv)*cHi + (z%zDiv)*cLo
//     off = base + (gm/rowsPerB)*cRowHi + (swap ? gn*ldc + gm%rowsPerB
//                                              : (gm%rowsPerB)*ldc + gn)
//   causal: if gn > gm -> -1e30 (after scale, before bias)
__global__ __launch_bounds__(256)
void gemm_bf16_wmma(const __bf16* __restrict__ A, long long sAz, int lda,
                    const __bf16* __restrict__ B, long long sBz, int ldb,
                    float* __restrict__ Cf, __bf16* __restrict__ Cb,
                    const float* __restrict__ RaddF, const __bf16* __restrict__ RaddB,
                    const float* __restrict__ bias,
                    int zDiv, long long cHi, long long cLo,
                    int ldc, int rowsPerB, long long cRowHi, int swapStore,
                    int M, int N, int K,
                    float scale, int relu, int causal)
{
  __shared__ __attribute__((aligned(16))) __bf16 As[2][BM * BKP];
  __shared__ __attribute__((aligned(16))) __bf16 Bs[2][BN * BKP];

  const int tid  = threadIdx.x;
  const int lane = tid & 31;
  const int wave = tid >> 5;
  const int wm   = wave & 3;    // 4 waves along M (32 rows each)
  const int wn   = wave >> 2;   // 2 waves along N (32 cols each)
  const int z    = blockIdx.z;
  const int rowBase = blockIdx.y * BM;
  const int colBase = blockIdx.x * BN;

  const __bf16* Ab = A + (long long)z * sAz;
  const __bf16* Bb = B + (long long)z * sBz;
  const long long cbase = (long long)(z / zDiv) * cHi + (long long)(z % zDiv) * cLo;

  // per-thread copy slots: A tile 128x32 = 512 16B-chunks (2/thread),
  //                        B tile  64x32 = 256 16B-chunks (1/thread)
  const int ch  = (tid & 3) * 8;          // k offset (bf16 elems) of 16B chunk
  const int ra0 = tid >> 2;               // A row 0..63
  const int ra1 = ra0 + 64;               // A row 64..127
  const int rb  = tid >> 2;               // B row 0..63
  const __bf16* gA0 = Ab + (long long)(rowBase + ra0) * lda + ch;
  const __bf16* gA1 = Ab + (long long)(rowBase + ra1) * lda + ch;
  const __bf16* gB0 = Bb + (long long)(colBase + rb ) * ldb + ch;
  __bf16* lA0[2] = { &As[0][ra0 * BKP + ch], &As[1][ra0 * BKP + ch] };
  __bf16* lA1[2] = { &As[0][ra1 * BKP + ch], &As[1][ra1 * BKP + ch] };
  __bf16* lB0[2] = { &Bs[0][rb  * BKP + ch], &Bs[1][rb  * BKP + ch] };

  v8f acc[2][2] = {};

  const int kb  = (lane >> 4) * 8;   // K sub-group per half-wave (bf16 layout)
  const int l16 = lane & 15;

  // prologue: stage tile 0 into buffer 0
  async_cp16(gA0, lA0[0]);
  async_cp16(gA1, lA1[0]);
  async_cp16(gB0, lB0[0]);

  const int KT = K / BK;
  for (int kt = 0; kt < KT; ++kt) {
    const int buf = kt & 1;
    if (kt + 1 < KT) {
      const int k1 = (kt + 1) * BK;
      async_cp16(gA0 + k1, lA0[buf ^ 1]);
      async_cp16(gA1 + k1, lA1[buf ^ 1]);
      async_cp16(gB0 + k1, lB0[buf ^ 1]);
      wait_async_pending();           // tile kt's 3 copies complete
    } else {
      wait_async_all();
    }
    __syncthreads();

    v16bf afrag[2], bfrag[2];
    for (int i = 0; i < 2; ++i) {
      int m = wm * 32 + i * 16 + l16;
      union { v16bf v; v8bf h[2]; } u;
      u.h[0] = *(const v8bf*)&As[buf][m * BKP + kb];
      u.h[1] = *(const v8bf*)&As[buf][m * BKP + kb + 16];
      afrag[i] = u.v;
    }
    for (int j = 0; j < 2; ++j) {
      int n = wn * 32 + j * 16 + l16;
      union { v16bf v; v8bf h[2]; } u;
      u.h[0] = *(const v8bf*)&Bs[buf][n * BKP + kb];
      u.h[1] = *(const v8bf*)&Bs[buf][n * BKP + kb + 16];
      bfrag[j] = u.v;
    }
    for (int i = 0; i < 2; ++i)
      for (int j = 0; j < 2; ++j)
        acc[i][j] = __builtin_amdgcn_wmma_f32_16x16x32_bf16(
            false, afrag[i], false, bfrag[j], (short)0, acc[i][j], false, false);

    __syncthreads();   // tile consumed; safe to re-stage this buffer
  }

  // epilogue
  for (int i = 0; i < 2; ++i) {
    for (int j = 0; j < 2; ++j) {
      const int gn = colBase + wn * 32 + j * 16 + l16;
      for (int r = 0; r < 8; ++r) {
        const int gm = rowBase + wm * 32 + i * 16 + ((lane < 16) ? r : r + 8);
        float c = acc[i][j][r] * scale;
        if (causal && gn > gm) c = -1e30f;
        if (bias) c += bias[gn];
        long long off = cbase + (long long)(gm / rowsPerB) * cRowHi;
        if (swapStore) off += (long long)gn * ldc + (gm % rowsPerB);
        else           off += (long long)(gm % rowsPerB) * ldc + gn;
        if (RaddF) c += RaddF[off];
        if (RaddB) c += (float)RaddB[off];
        if (relu) c = fmaxf(c, 0.f);
        if (Cf) Cf[off] = c;
        if (Cb) Cb[off] = cvt_bf16(c);
      }
    }
  }
}

// ---- transpose + convert: in f32 [K,N] -> out bf16 [N,K], batched ----------
__global__ __launch_bounds__(256)
void transpose_cvt_k(const float* __restrict__ in, __bf16* __restrict__ outT,
                     int K, int N)
{
  __shared__ float tile[32][33];
  const long long z = blockIdx.z;
  const float* inb = in + z * (long long)K * N;
  __bf16* outb = outT + z * (long long)K * N;
  const int n0 = blockIdx.x * 32, k0 = blockIdx.y * 32;
  const int tx = threadIdx.x, ty = threadIdx.y;   // block (32,8)
  for (int j = 0; j < 4; ++j)
    tile[ty + j * 8][tx] = inb[(long long)(k0 + ty + j * 8) * N + (n0 + tx)];
  __syncthreads();
  for (int j = 0; j < 4; ++j)
    outb[(long long)(n0 + ty + j * 8) * K + (k0 + tx)] = cvt_bf16(tile[tx][ty + j * 8]);
}

// ---- LayerNorm (f32 in, bf16 out): one block per row of length E -----------
__global__ __launch_bounds__(256)
void layernorm_k(const float* __restrict__ x, const float* __restrict__ g,
                 const float* __restrict__ b, __bf16* __restrict__ y, int E)
{
  __shared__ float red[8];
  __shared__ float stat[2];
  const long long row = blockIdx.x;
  const float* xr = x + row * E;
  __bf16* yr = y + row * E;
  const int tid = threadIdx.x;

  float s = 0.f;
  for (int i = tid; i < E; i += 256) s += xr[i];
  for (int o = 16; o > 0; o >>= 1) s += __shfl_down(s, o, 32);
  if ((tid & 31) == 0) red[tid >> 5] = s;
  __syncthreads();
  if (tid == 0) { float t = 0.f; for (int i = 0; i < 8; ++i) t += red[i]; stat[0] = t / E; }
  __syncthreads();
  const float mean = stat[0];

  float v = 0.f;
  for (int i = tid; i < E; i += 256) { float d = xr[i] - mean; v += d * d; }
  for (int o = 16; o > 0; o >>= 1) v += __shfl_down(v, o, 32);
  __syncthreads();
  if ((tid & 31) == 0) red[tid >> 5] = v;
  __syncthreads();
  if (tid == 0) { float t = 0.f; for (int i = 0; i < 8; ++i) t += red[i];
                  stat[1] = rsqrtf(t / E + 1e-5f); }
  __syncthreads();
  const float rstd = stat[1];
  for (int i = tid; i < E; i += 256)
    yr[i] = cvt_bf16((xr[i] - mean) * rstd * g[i] + b[i]);
}

// ---- Softmax (f32 in, bf16 out): one block per row -------------------------
__global__ __launch_bounds__(256)
void softmax_bf_k(const float* __restrict__ S, __bf16* __restrict__ O, int T)
{
  __shared__ float red[8];
  __shared__ float stat;
  const long long row = blockIdx.x;
  const float* sr = S + row * (long long)T;
  __bf16* orow = O + row * (long long)T;
  const int tid = threadIdx.x;

  float m = -3.0e38f;
  for (int i = tid; i < T; i += 256) m = fmaxf(m, sr[i]);
  for (int o = 16; o > 0; o >>= 1) m = fmaxf(m, __shfl_down(m, o, 32));
  if ((tid & 31) == 0) red[tid >> 5] = m;
  __syncthreads();
  if (tid == 0) { float t = red[0]; for (int i = 1; i < 8; ++i) t = fmaxf(t, red[i]); stat = t; }
  __syncthreads();
  const float mx = stat;

  float sum = 0.f;
  for (int i = tid; i < T; i += 256) sum += __expf(sr[i] - mx);
  for (int o = 16; o > 0; o >>= 1) sum += __shfl_down(sum, o, 32);
  __syncthreads();
  if ((tid & 31) == 0) red[tid >> 5] = sum;
  __syncthreads();
  if (tid == 0) { float t = 0.f; for (int i = 0; i < 8; ++i) t += red[i]; stat = 1.0f / t; }
  __syncthreads();
  const float inv = stat;
  for (int i = tid; i < T; i += 256)
    orow[i] = cvt_bf16(__expf(sr[i] - mx) * inv);
}

// ---- Embedding gather: x[b,t,:] = tok_emb[idx[b,t],:] + pos_emb[t,:] -------
__global__ __launch_bounds__(256)
void embed_k(const int* __restrict__ idx, const float* __restrict__ tok,
             const float* __restrict__ pos, float* __restrict__ x,
             int T, int E, long long total)
{
  long long i = (long long)blockIdx.x * 256 + threadIdx.x;
  if (i >= total) return;
  int e = (int)(i % E);
  long long bt = i / E;
  int t = (int)(bt % T);
  int token = idx[bt];
  x[i] = tok[(long long)token * E + e] + pos[(long long)t * E + e];
}

// ---------------------------------------------------------------------------
extern "C" void kernel_launch(void* const* d_in, const int* in_sizes, int n_in,
                              void* d_out, int out_size, void* d_ws, size_t ws_size,
                              hipStream_t stream) {
  (void)in_sizes; (void)n_in; (void)out_size; (void)ws_size;

  const int Vv = 32000, E = 1024, H = 16, L = 8, HS = 64, Bd = 2, T = 1024;
  const int M = Bd * T;          // 2048 token rows
  const int FF = 4 * E;          // 4096
  const int BIGDIV = 1 << 30;

  const int*   idx   = (const int*)  d_in[0];
  const float* tok   = (const float*)d_in[1];
  const float* pos   = (const float*)d_in[2];
  const float* ln1_g = (const float*)d_in[3];
  const float* ln1_b = (const float*)d_in[4];
  const float* wq    = (const float*)d_in[5];
  const float* wk    = (const float*)d_in[6];
  const float* wv    = (const float*)d_in[7];
  const float* wproj = (const float*)d_in[8];
  const float* bproj = (const float*)d_in[9];
  const float* ln2_g = (const float*)d_in[10];
  const float* ln2_b = (const float*)d_in[11];
  const float* w1    = (const float*)d_in[12];
  const float* b1    = (const float*)d_in[13];
  const float* w2    = (const float*)d_in[14];
  const float* b2    = (const float*)d_in[15];
  const float* lnf_g = (const float*)d_in[16];
  const float* lnf_b = (const float*)d_in[17];
  const float* wfc   = (const float*)d_in[18];
  const float* bfc   = (const float*)d_in[19];
  float* out = (float*)d_out;

  // ---- workspace layout ----
  const size_t ME  = (size_t)M * E;                    // 2M
  const size_t SCE = (size_t)Bd * H * T * T;           // 33.5M
  float* X  = (float*)d_ws;          // [B,T,E] residual stream (f32)
  float* X1 = X + ME;                // [B,T,E] post-attn residual (f32)
  float* Sc = X1 + ME;               // [B,H,T,T] raw scores (f32)
  __bf16* bp   = (__bf16*)(Sc + SCE);
  __bf16* Hbf  = bp;            bp += ME;              // LN out (h / h2 / final)
  __bf16* Qbf  = bp;            bp += ME;              // [B,H,T,HS]
  __bf16* Kbf  = bp;            bp += ME;              // [B,H,T,HS]
  __bf16* VTbf = bp;            bp += ME;              // [B,H,HS,T]
  __bf16* Obf  = bp;            bp += ME;              // [B,T,E]
  __bf16* F1bf = bp;            bp += (size_t)M * FF;  // [B,T,4E]
  __bf16* Atbf = bp;            bp += SCE;             // softmaxed attn
  __bf16* WqT  = bp;            bp += (size_t)L * H * E * HS;  // [L,H,HS,E]
  __bf16* WkT  = bp;            bp += (size_t)L * H * E * HS;
  __bf16* WvT  = bp;            bp += (size_t)L * H * E * HS;
  __bf16* WpT  = bp;            bp += (size_t)L * E * E;       // [L,E,E]
  __bf16* W1T  = bp;            bp += (size_t)L * E * FF;      // [L,FF,E]
  __bf16* W2T  = bp;            bp += (size_t)L * FF * E;      // [L,E,FF]
  __bf16* WfcT = bp;                                            // [V,E]

  // ---- pre-pass: weights f32 [K,N] -> bf16 [N,K] ----
  {
    dim3 tb(32, 8);
    dim3 g_qkv(HS / 32, E / 32, L * H);
    transpose_cvt_k<<<g_qkv, tb, 0, stream>>>(wq, WqT, E, HS);
    transpose_cvt_k<<<g_qkv, tb, 0, stream>>>(wk, WkT, E, HS);
    transpose_cvt_k<<<g_qkv, tb, 0, stream>>>(wv, WvT, E, HS);
    dim3 g_p(E / 32, E / 32, L);
    transpose_cvt_k<<<g_p, tb, 0, stream>>>(wproj, WpT, E, E);
    dim3 g_1(FF / 32, E / 32, L);
    transpose_cvt_k<<<g_1, tb, 0, stream>>>(w1, W1T, E, FF);
    dim3 g_2(E / 32, FF / 32, L);
    transpose_cvt_k<<<g_2, tb, 0, stream>>>(w2, W2T, FF, E);
    dim3 g_f(Vv / 32, E / 32, 1);
    transpose_cvt_k<<<g_f, tb, 0, stream>>>(wfc, WfcT, E, Vv);
  }

  // x = tok_emb[idx] + pos_emb
  {
    long long total = (long long)M * E;
    embed_k<<<(unsigned)((total + 255) / 256), 256, 0, stream>>>(idx, tok, pos, X, T, E, total);
  }

  for (int l = 0; l < L; ++l) {
    // h = ln1(x) -> bf16
    layernorm_k<<<M, 256, 0, stream>>>(X, ln1_g + (size_t)l * E, ln1_b + (size_t)l * E, Hbf, E);

    // q/k[b,h,t,d] and vT[b,h,d,t] (grid.z = head)
    dim3 g_qkv(1, M / BM, H);
    gemm_bf16_wmma<<<g_qkv, 256, 0, stream>>>(
        Hbf, 0LL, E,  WqT + (size_t)l * H * E * HS, (long long)E * HS, E,
        nullptr, Qbf, nullptr, nullptr, nullptr,
        BIGDIV, 0LL, (long long)T * HS, HS, T, (long long)H * T * HS, 0,
        M, HS, E, 1.0f, 0, 0);
    gemm_bf16_wmma<<<g_qkv, 256, 0, stream>>>(
        Hbf, 0LL, E,  WkT + (size_t)l * H * E * HS, (long long)E * HS, E,
        nullptr, Kbf, nullptr, nullptr, nullptr,
        BIGDIV, 0LL, (long long)T * HS, HS, T, (long long)H * T * HS, 0,
        M, HS, E, 1.0f, 0, 0);
    gemm_bf16_wmma<<<g_qkv, 256, 0, stream>>>(   // vT: swapped store
        Hbf, 0LL, E,  WvT + (size_t)l * H * E * HS, (long long)E * HS, E,
        nullptr, VTbf, nullptr, nullptr, nullptr,
        BIGDIV, 0LL, (long long)HS * T, T, T, (long long)H * HS * T, 1,
        M, HS, E, 1.0f, 0, 0);

    // scores[b,h,t,s] = (k[t] . q[s]) * HS^-0.5, causal (s <= t kept) -> f32
    dim3 g_sc(T / BN, T / BM, Bd * H);
    gemm_bf16_wmma<<<g_sc, 256, 0, stream>>>(
        Kbf, (long long)T * HS, HS,  Qbf, (long long)T * HS, HS,
        Sc, nullptr, nullptr, nullptr, nullptr,
        BIGDIV, 0LL, (long long)T * T, T, T, 0LL, 0,
        T, T, HS, 0.125f, 0, 1);

    // softmax over s -> bf16
    softmax_bf_k<<<Bd * H * T, 256, 0, stream>>>(Sc, Atbf, T);

    // o[b,t,h*HS+d] = attn @ v   (B operand = vT [HS,T])
    dim3 g_av(1, T / BM, Bd * H);
    gemm_bf16_wmma<<<g_av, 256, 0, stream>>>(
        Atbf, (long long)T * T, T,  VTbf, (long long)HS * T, T,
        nullptr, Obf, nullptr, nullptr, nullptr,
        H, (long long)T * E, (long long)HS, E, T, 0LL, 0,
        T, HS, T, 1.0f, 0, 0);

    // x1 = h + o @ wproj + bproj  (f32 residual; h added as bf16 residual)
    dim3 g_pr(E / BN, M / BM, 1);
    gemm_bf16_wmma<<<g_pr, 256, 0, stream>>>(
        Obf, 0LL, E,  WpT + (size_t)l * E * E, 0LL, E,
        X1, nullptr, nullptr, Hbf, bproj + (size_t)l * E,
        BIGDIV, 0LL, 0LL, E, M, 0LL, 0,
        M, E, E, 1.0f, 0, 0);

    // h2 = ln2(x1) -> bf16 (reuse Hbf)
    layernorm_k<<<M, 256, 0, stream>>>(X1, ln2_g + (size_t)l * E, ln2_b + (size_t)l * E, Hbf, E);

    // f1 = relu(h2 @ w1 + b1) -> bf16
    dim3 g_f1(FF / BN, M / BM, 1);
    gemm_bf16_wmma<<<g_f1, 256, 0, stream>>>(
        Hbf, 0LL, E,  W1T + (size_t)l * E * FF, 0LL, E,
        nullptr, F1bf, nullptr, nullptr, b1 + (size_t)l * FF,
        BIGDIV, 0LL, 0LL, FF, M, 0LL, 0,
        M, FF, E, 1.0f, 1, 0);

    // x = x1 + f1 @ w2 + b2  (f32)
    dim3 g_f2(E / BN, M / BM, 1);
    gemm_bf16_wmma<<<g_f2, 256, 0, stream>>>(
        F1bf, 0LL, FF,  W2T + (size_t)l * FF * E, 0LL, FF,
        X, nullptr, X1, nullptr, b2 + (size_t)l * E,
        BIGDIV, 0LL, 0LL, E, M, 0LL, 0,
        M, E, FF, 1.0f, 0, 0);
  }

  // final LN + vocab projection
  layernorm_k<<<M, 256, 0, stream>>>(X, lnf_g, lnf_b, Hbf, E);
  dim3 g_lg(Vv / BN, M / BM, 1);
  gemm_bf16_wmma<<<g_lg, 256, 0, stream>>>(
      Hbf, 0LL, E,  WfcT, 0LL, E,
      out, nullptr, nullptr, nullptr, bfc,
      BIGDIV, 0LL, 0LL, Vv, M, 0LL, 0,
      M, Vv, E, 1.0f, 0, 0);
}
